// TheoremProver_37864431681879
// MI455X (gfx1250) — compile-verified, matrix operand order checked by reading, not run on hardware
//
#include <hip/hip_runtime.h>
#include <math.h>

// ---------------- constants (match reference) ----------------
constexpr int cB   = 64;
constexpr int cH   = 1024;
constexpr int cNH  = 8;
constexpr int cDH  = 128;     // cH / cNH
constexpr int cS   = 16384;   // S_AX
constexpr int cSTEPS = 16;
constexpr int cSPLITS = 64;   // flash-attention S splits
constexpr int cCHUNK  = cS / cSPLITS;   // 256 keys per split
constexpr float cSCALE = 0.08838834764831845f; // 1/sqrt(128)

typedef unsigned short u16;
typedef __attribute__((ext_vector_type(16))) __bf16        v16bf;
typedef __attribute__((ext_vector_type(16))) unsigned short v16us;
typedef __attribute__((ext_vector_type(8)))  unsigned short v8us;
typedef __attribute__((ext_vector_type(8)))  float          v8f;
typedef int tp_v4i __attribute__((__vector_size__(4 * sizeof(int)))); // builtin's pointee type

// gfx1250 async global->LDS DMA (ASYNCcnt-tracked). Guarded: host pass and
// toolchains without the builtin fall back to the plain load path.
#if defined(__gfx1250__) && __has_builtin(__builtin_amdgcn_global_load_async_to_lds_b128) && \
    __has_builtin(__builtin_amdgcn_s_wait_asynccnt)
#define TP_HAS_ASYNC 1
#else
#define TP_HAS_ASYNC 0
#endif
#define TP_AS1 __attribute__((address_space(1)))
#define TP_AS3 __attribute__((address_space(3)))

// ---------------- bf16 helpers (bit-level, no __bf16 arithmetic) ----------------
__device__ __forceinline__ u16 f2bf(float f) {
  union { float f; unsigned int u; } v; v.f = f;
  unsigned int r = v.u + 0x7fffu + ((v.u >> 16) & 1u);  // RNE
  return (u16)(r >> 16);
}
__device__ __forceinline__ float bf2f(u16 h) {
  union { unsigned int u; float f; } v; v.u = ((unsigned int)h) << 16;
  return v.f;
}

// Build a 16x32 bf16 A/B WMMA fragment for this lane.
// Layout: lanes 0-15 hold K {kk..kk+7} and {kk+16..kk+23};
//         lanes 16-31 hold K {kk+8..kk+15} and {kk+24..kk+31}. (ko = (lane>=16)?8:0)
__device__ __forceinline__ v16bf ldfrag(const u16* p, int kk, int ko) {
  v8us lo = *(const v8us*)(p + kk + ko);
  v8us hi = *(const v8us*)(p + kk + ko + 16);
  v16us u = __builtin_shufflevector(lo, hi, 0,1,2,3,4,5,6,7,8,9,10,11,12,13,14,15);
  union { v16us u; v16bf b; } cv; cv.u = u;
  return cv.b;
}

__device__ __forceinline__ v8f wmma_bf16(v16bf a, v16bf b, v8f c) {
  return __builtin_amdgcn_wmma_f32_16x16x32_bf16(false, a, false, b, (short)0, c, false, false);
}

__device__ __forceinline__ float half16_max(float v) {
  v = fmaxf(v, __shfl_xor(v, 1, 32));
  v = fmaxf(v, __shfl_xor(v, 2, 32));
  v = fmaxf(v, __shfl_xor(v, 4, 32));
  v = fmaxf(v, __shfl_xor(v, 8, 32));
  return v;
}
__device__ __forceinline__ float half16_sum(float v) {
  v += __shfl_xor(v, 1, 32);
  v += __shfl_xor(v, 2, 32);
  v += __shfl_xor(v, 4, 32);
  v += __shfl_xor(v, 8, 32);
  return v;
}

// ---------------- elementwise kernels ----------------
__global__ void tp_cast_f2bf(const float* __restrict__ x, u16* __restrict__ y, int n) {
  for (int i = blockIdx.x * blockDim.x + threadIdx.x; i < n; i += gridDim.x * blockDim.x)
    y[i] = f2bf(x[i]);
}
__global__ void tp_copy_f32(const float* __restrict__ x, float* __restrict__ y, int n) {
  for (int i = blockIdx.x * blockDim.x + threadIdx.x; i < n; i += gridDim.x * blockDim.x)
    y[i] = x[i];
}
__global__ void tp_copy_u16_strided(const u16* __restrict__ s, u16* __restrict__ d,
                                    int rows, int cols, int ldd) {
  int n = rows * cols;
  for (int i = blockIdx.x * blockDim.x + threadIdx.x; i < n; i += gridDim.x * blockDim.x)
    d[(i / cols) * ldd + (i % cols)] = s[i];
}

// ---------------- WMMA GEMM: out = act((A[M,K] . W[N,K]^T + bias) * scale) ----------------
// Block = 256 threads = 8 waves; each wave computes a 16x64 output tile (4 accumulators reuse A).
__global__ __launch_bounds__(256) void tp_wmma_gemm(
    const u16* __restrict__ A, const u16* __restrict__ W, const float* __restrict__ bias,
    float* __restrict__ Cf, int ldcf, u16* __restrict__ Cb, int ldcb,
    int M, int N, int K, float scale, int act) {
  const int lane = threadIdx.x & 31;
  const int wave = threadIdx.x >> 5;
  const int m0 = blockIdx.y * 16;
  const int n0 = blockIdx.x * 512 + wave * 64;
  if (m0 >= M || n0 >= N) return;
  const int row = lane & 15;
  const int ko  = (lane >> 4) * 8;

  const u16* arow = A + (size_t)(m0 + row) * K;
  const u16* w0 = W + (size_t)(n0 +  0 + row) * K;
  const u16* w1 = W + (size_t)(n0 + 16 + row) * K;
  const u16* w2 = W + (size_t)(n0 + 32 + row) * K;
  const u16* w3 = W + (size_t)(n0 + 48 + row) * K;

  v8f acc0 = {}, acc1 = {}, acc2 = {}, acc3 = {};
  for (int kk = 0; kk < K; kk += 32) {
    v16bf a = ldfrag(arow, kk, ko);
    acc0 = wmma_bf16(a, ldfrag(w0, kk, ko), acc0);
    acc1 = wmma_bf16(a, ldfrag(w1, kk, ko), acc1);
    acc2 = wmma_bf16(a, ldfrag(w2, kk, ko), acc2);
    acc3 = wmma_bf16(a, ldfrag(w3, kk, ko), acc3);
  }

  const int mrow = m0 + ((lane >> 4) << 3);   // +r below
  const int ncol = lane & 15;
#pragma unroll
  for (int t = 0; t < 4; t++) {
    v8f acc = (t == 0) ? acc0 : (t == 1) ? acc1 : (t == 2) ? acc2 : acc3;
    int n = n0 + t * 16 + ncol;
    float bv = bias ? bias[n] : 0.0f;
#pragma unroll
    for (int r = 0; r < 8; r++) {
      int m = mrow + r;
      float v = (acc[r] + bv) * scale;
      if (act == 1) v = 0.5f * v * (1.0f + erff(v * 0.70710678118654752f)); // exact GELU
      if (Cf) Cf[(size_t)m * ldcf + n] = v;
      if (Cb) Cb[(size_t)m * ldcb + n] = f2bf(v);
    }
  }
}

// ---------------- flash attention over axioms (split-S, partials) ----------------
// grid = (cSPLITS, cNH, cB/16), block = 32 (one wave).
// q[B, H] bf16 (pre-scaled); Kp/Vp [S, H] bf16 (head-major inside row).
// K tile is staged global->LDS with ASYNCcnt-tracked async DMA (gfx1250);
// V tile goes through VGPRs because it is transposed on the way into LDS.
// Writes Opart[sp, B, NH, DH] (f32, unnormalized), Mpart/Lpart [sp, B, NH].
__global__ __launch_bounds__(32) void tp_flash_axiom(
    const u16* __restrict__ q, const u16* __restrict__ Kp, const u16* __restrict__ Vp,
    float* __restrict__ Op, float* __restrict__ Mp, float* __restrict__ Lp) {
  __shared__ u16 lK[32 * 128];   // [key][feat]
  __shared__ u16 lVT[128 * 32];  // [feat][key]
  __shared__ u16 lP[16 * 32];    // [row][key]
  const int lane = threadIdx.x;
  const int sp = blockIdx.x, h = blockIdx.y, b0 = blockIdx.z * 16;
  const int row = lane & 15;
  const int ko  = (lane >> 4) * 8;

  // Q fragments (loop invariant): 16 rows x 128 features -> 4 K-slices
  const u16* qrow = q + (size_t)(b0 + row) * cH + h * cDH;
  v16bf qf[4];
#pragma unroll
  for (int i = 0; i < 4; i++) qf[i] = ldfrag(qrow, i * 32, ko);

  v8f acc[8]; // O accumulator: 8 feature tiles of 16
#pragma unroll
  for (int i = 0; i < 8; i++) acc[i] = (v8f){};
  float rm[8], rl[8];
#pragma unroll
  for (int r = 0; r < 8; r++) { rm[r] = -1e30f; rl[r] = 0.0f; }

  const int sbeg = sp * cCHUNK;
  for (int s0 = sbeg; s0 < sbeg + cCHUNK; s0 += 32) {
    __syncthreads();   // previous tile's LDS readers are done

#if TP_HAS_ASYNC
    // K tile: async DMA straight into LDS (no VGPR round-trip, ASYNCcnt-tracked)
#pragma unroll
    for (int i = 0; i < 16; i++) {
      int idx = i * 32 + lane;
      int key = idx >> 4;
      int f0  = (idx & 15) * 8;
      size_t g = (size_t)(s0 + key) * cH + h * cDH + f0;
      __builtin_amdgcn_global_load_async_to_lds_b128(
          (TP_AS1 tp_v4i*)(u16*)(Kp + g),
          (TP_AS3 tp_v4i*)&lK[key * 128 + f0], 0, 0);
    }
#endif
    // V tile: load to VGPRs, store transposed into LDS (overlaps the async K DMA)
#pragma unroll
    for (int i = 0; i < 16; i++) {
      int idx = i * 32 + lane;
      int key = idx >> 4;
      int f0  = (idx & 15) * 8;
      size_t g = (size_t)(s0 + key) * cH + h * cDH + f0;
#if !TP_HAS_ASYNC
      *(v8us*)&lK[key * 128 + f0] = *(const v8us*)(Kp + g);
#endif
      v8us vv = *(const v8us*)(Vp + g);
#pragma unroll
      for (int e = 0; e < 8; e++) lVT[(f0 + e) * 32 + key] = vv[e];
    }
    // keep the L2->LDS stream ahead: prefetch next chunk's K/V rows
    if (s0 + 32 < sbeg + cCHUNK) {
      size_t gp = (size_t)(s0 + 32 + (lane >> 1)) * cH + h * cDH + (lane & 1) * 64;
      __builtin_prefetch((const void*)(Kp + gp), 0, 0);
      __builtin_prefetch((const void*)(Vp + gp), 0, 0);
    }
#if TP_HAS_ASYNC
    __builtin_amdgcn_s_wait_asynccnt(0);
#endif
    __syncthreads();

    // scores S[16x32] = Q . K^T  (two 16-key column tiles)
    v8f sc[2];
#pragma unroll
    for (int nt = 0; nt < 2; nt++) {
      v8f s = {};
      const u16* krow = &lK[(nt * 16 + row) * 128];
#pragma unroll
      for (int kk = 0; kk < 128; kk += 32)
        s = wmma_bf16(qf[kk >> 5], ldfrag(krow, kk, ko), s);
      sc[nt] = s;
    }

    // online softmax; register r holds rows r (lanes 0-15) and 8+r (lanes 16-31)
#pragma unroll
    for (int r = 0; r < 8; r++) {
      float t0 = sc[0][r], t1 = sc[1][r];
      float tm = half16_max(fmaxf(t0, t1));
      float nm = fmaxf(rm[r], tm);
      float corr = __expf(rm[r] - nm);
      float p0 = __expf(t0 - nm), p1 = __expf(t1 - nm);
      rl[r] = rl[r] * corr + half16_sum(p0) + half16_sum(p1);
      rm[r] = nm;
#pragma unroll
      for (int nt2 = 0; nt2 < 8; nt2++) acc[nt2][r] *= corr;
      int prow = r + ((lane >> 4) << 3);
      lP[prow * 32 + (lane & 15)]      = f2bf(p0);
      lP[prow * 32 + 16 + (lane & 15)] = f2bf(p1);
    }
    __syncthreads();

    // O += P[16x32] . V[32x128]
    v16bf pa = ldfrag(&lP[row * 32], 0, ko);
#pragma unroll
    for (int nt2 = 0; nt2 < 8; nt2++) {
      const u16* vrow = &lVT[(nt2 * 16 + row) * 32];
      acc[nt2] = wmma_bf16(pa, ldfrag(vrow, 0, ko), acc[nt2]);
    }
  }

  // write partials (unnormalized)
#pragma unroll
  for (int nt2 = 0; nt2 < 8; nt2++)
#pragma unroll
    for (int r = 0; r < 8; r++) {
      int m = r + ((lane >> 4) << 3);
      int f = nt2 * 16 + (lane & 15);
      Op[(((size_t)sp * cB + b0 + m) * cNH + h) * cDH + f] = acc[nt2][r];
    }
  if ((lane & 15) == 0) {
#pragma unroll
    for (int r = 0; r < 8; r++) {
      int m = r + ((lane >> 4) << 3);
      size_t ix = ((size_t)sp * cB + b0 + m) * cNH + h;
      Mp[ix] = rm[r];
      Lp[ix] = rl[r];
    }
  }
}

// combine split partials -> bf16 attended vectors [B, H]
__global__ __launch_bounds__(128) void tp_flash_combine(
    const float* __restrict__ Op, const float* __restrict__ Mp, const float* __restrict__ Lp,
    u16* __restrict__ out) {
  int b = blockIdx.x, h = blockIdx.y, d = threadIdx.x;
  float mx = -1e30f;
  for (int sp = 0; sp < cSPLITS; sp++)
    mx = fmaxf(mx, Mp[((size_t)sp * cB + b) * cNH + h]);
  float L = 0.0f, o = 0.0f;
  for (int sp = 0; sp < cSPLITS; sp++) {
    size_t ix = ((size_t)sp * cB + b) * cNH + h;
    float w = __expf(Mp[ix] - mx);
    L += Lp[ix] * w;
    o += Op[ix * cDH + d] * w;
  }
  out[(size_t)b * cH + h * cDH + d] = f2bf(o / L);
}

// ---------------- tiny attention over proof steps (T <= 16) ----------------
__global__ __launch_bounds__(128) void tp_step_attn(
    const u16* __restrict__ qs, const u16* __restrict__ kc, const u16* __restrict__ vc,
    int T, u16* __restrict__ out) {
  __shared__ float red[128];
  __shared__ float sc[32];
  int b = blockIdx.x, h = blockIdx.y, d = threadIdx.x;
  size_t off = (size_t)b * cH + h * cDH + d;
  float qv = bf2f(qs[off]);
  for (int t = 0; t < T; t++) {
    red[d] = qv * bf2f(kc[(size_t)t * cB * cH + off]);
    __syncthreads();
    for (int s = 64; s > 0; s >>= 1) { if (d < s) red[d] += red[d + s]; __syncthreads(); }
    if (d == 0) sc[t] = red[0];
    __syncthreads();
  }
  float mx = -1e30f;
  for (int t = 0; t < T; t++) mx = fmaxf(mx, sc[t]);
  float sum = 0.0f, o = 0.0f;
  for (int t = 0; t < T; t++) {
    float pp = __expf(sc[t] - mx);
    sum += pp;
    o += pp * bf2f(vc[(size_t)t * cB * cH + off]);
  }
  out[off] = f2bf(o / sum);
}

// ---------------- LayerNorm + exact GELU over rows of 2048, fp32 in -> bf16 out ----------------
__global__ __launch_bounds__(256) void tp_ln_gelu(
    const float* __restrict__ X, const float* __restrict__ g, const float* __restrict__ bb,
    u16* __restrict__ Y) {
  __shared__ float red[256];
  __shared__ float sh_mean, sh_inv;
  int r = blockIdx.x, t = threadIdx.x;
  const float* x = X + (size_t)r * 2048;
  float s = 0.0f;
  for (int i = t; i < 2048; i += 256) s += x[i];
  red[t] = s; __syncthreads();
  for (int k = 128; k > 0; k >>= 1) { if (t < k) red[t] += red[t + k]; __syncthreads(); }
  if (t == 0) sh_mean = red[0] * (1.0f / 2048.0f);
  __syncthreads();
  float mean = sh_mean;
  s = 0.0f;
  for (int i = t; i < 2048; i += 256) { float d = x[i] - mean; s += d * d; }
  red[t] = s; __syncthreads();
  for (int k = 128; k > 0; k >>= 1) { if (t < k) red[t] += red[t + k]; __syncthreads(); }
  if (t == 0) sh_inv = rsqrtf(red[0] * (1.0f / 2048.0f) + 1e-5f);
  __syncthreads();
  float inv = sh_inv;
  for (int i = t; i < 2048; i += 256) {
    float y = (x[i] - mean) * inv * g[i] + bb[i];
    y = 0.5f * y * (1.0f + erff(y * 0.70710678118654752f));
    Y[(size_t)r * 2048 + i] = f2bf(y);
  }
}

// ---------------- validity head: sigmoid(gelu_h . w2 + b2) ----------------
__global__ __launch_bounds__(256) void tp_val_head(
    const u16* __restrict__ Hh, const float* __restrict__ w2, const float* __restrict__ b2,
    float* __restrict__ out) {
  __shared__ float red[256];
  int b = blockIdx.x, t = threadIdx.x;
  float s = 0.0f;
  for (int i = t; i < cH; i += 256) s += bf2f(Hh[(size_t)b * cH + i]) * w2[i];
  red[t] = s; __syncthreads();
  for (int k = 128; k > 0; k >>= 1) { if (t < k) red[t] += red[t + k]; __syncthreads(); }
  if (t == 0) out[b] = 1.0f / (1.0f + __expf(-(red[0] + b2[0])));
}

// ---------------- host orchestration ----------------
static void gemm(hipStream_t s, const u16* A, const u16* W, const float* bias,
                 float* Cf, int ldcf, u16* Cb, int ldcb,
                 int M, int N, int K, float scale, int act) {
  dim3 g((N + 511) / 512, (M + 15) / 16);
  tp_wmma_gemm<<<g, 256, 0, s>>>(A, W, bias, Cf, ldcf, Cb, ldcb, M, N, K, scale, act);
}

extern "C" void kernel_launch(void* const* d_in, const int* in_sizes, int n_in,
                              void* d_out, int out_size, void* d_ws, size_t ws_size,
                              hipStream_t stream) {
  (void)in_sizes; (void)n_in; (void)ws_size; (void)out_size;
  const float* conj   = (const float*)d_in[0];
  const float* axemb  = (const float*)d_in[1];
  const float* axinw  = (const float*)d_in[2];
  const float* axinb  = (const float*)d_in[3];
  const float* axoutw = (const float*)d_in[4];
  const float* axoutb = (const float*)d_in[5];
  const float* stinw  = (const float*)d_in[6];
  const float* stinb  = (const float*)d_in[7];
  const float* stoutw = (const float*)d_in[8];
  const float* stoutb = (const float*)d_in[9];
  const float* genw1  = (const float*)d_in[10];
  const float* genb1  = (const float*)d_in[11];
  const float* lng    = (const float*)d_in[12];
  const float* lnb    = (const float*)d_in[13];
  const float* genw2  = (const float*)d_in[14];
  const float* genb2  = (const float*)d_in[15];
  const float* valw1  = (const float*)d_in[16];
  const float* valb1  = (const float*)d_in[17];
  const float* valw2  = (const float*)d_in[18];
  const float* valb2  = (const float*)d_in[19];
  // d_in[20..27] = LSTM params: provably dead w.r.t. (steps, val) outputs -> eliminated.
  // d_in[28] = max_steps = 16 (fixed reference constant).

  const int HH = cH * cH;        // 1M
  const int BH = cB * cH;        // 64K
  float* outp = (float*)d_out;

  // workspace carve-up
  char* p = (char*)d_ws;
  auto alloc = [&](size_t bytes) -> void* {
    void* r = (void*)p; p += (bytes + 255) & ~(size_t)255; return r;
  };
  u16* axinw_b  = (u16*)alloc((size_t)3 * HH * 2);
  u16* stinw_b  = (u16*)alloc((size_t)3 * HH * 2);
  u16* axoutw_b = (u16*)alloc((size_t)HH * 2);
  u16* stoutw_b = (u16*)alloc((size_t)HH * 2);
  u16* genw1_b  = (u16*)alloc((size_t)2048 * 3072 * 2);
  u16* genw2_b  = (u16*)alloc((size_t)1024 * 2048 * 2);
  u16* valw1_b  = (u16*)alloc((size_t)HH * 2);
  u16* axemb_b  = (u16*)alloc((size_t)cS * cH * 2);
  u16* kpa      = (u16*)alloc((size_t)cS * cH * 2);
  u16* vpa      = (u16*)alloc((size_t)cS * cH * 2);
  u16* q_b      = (u16*)alloc((size_t)BH * 2);
  u16* qs_b     = (u16*)alloc((size_t)BH * 2);
  u16* kcache   = (u16*)alloc((size_t)(cSTEPS + 1) * BH * 2);
  u16* vcache   = (u16*)alloc((size_t)(cSTEPS + 1) * BH * 2);
  u16* attn_b   = (u16*)alloc((size_t)BH * 2);
  u16* statt_b  = (u16*)alloc((size_t)BH * 2);
  u16* comb_b   = (u16*)alloc((size_t)cB * 3072 * 2);
  u16* state_b  = (u16*)alloc((size_t)BH * 2);
  u16* x_b      = (u16*)alloc((size_t)cB * 2048 * 2);
  u16* valh_b   = (u16*)alloc((size_t)BH * 2);
  float* x_f    = (float*)alloc((size_t)cB * 2048 * 4);
  float* Opart  = (float*)alloc((size_t)cSPLITS * cB * cNH * cDH * 4);
  float* Mpart  = (float*)alloc((size_t)cSPLITS * cB * cNH * 4);
  float* Lpart  = (float*)alloc((size_t)cSPLITS * cB * cNH * 4);

  // 1) convert weights + axiom table to bf16
  tp_cast_f2bf<<<2048, 256, 0, stream>>>(axinw,  axinw_b,  3 * HH);
  tp_cast_f2bf<<<2048, 256, 0, stream>>>(stinw,  stinw_b,  3 * HH);
  tp_cast_f2bf<<<1024, 256, 0, stream>>>(axoutw, axoutw_b, HH);
  tp_cast_f2bf<<<1024, 256, 0, stream>>>(stoutw, stoutw_b, HH);
  tp_cast_f2bf<<<2048, 256, 0, stream>>>(genw1,  genw1_b,  2048 * 3072);
  tp_cast_f2bf<<<1024, 256, 0, stream>>>(genw2,  genw2_b,  1024 * 2048);
  tp_cast_f2bf<<<1024, 256, 0, stream>>>(valw1,  valw1_b,  HH);
  tp_cast_f2bf<<<4096, 256, 0, stream>>>(axemb,  axemb_b,  cS * cH);

  // 2) one-time axiom K/V projection (bf16 results stay L2-resident: 64 MB << 192 MB L2)
  gemm(stream, axemb_b, axinw_b + (size_t)HH,     axinb + cH,     nullptr, 0, kpa, cH, cS, cH, cH, 1.0f, 0);
  gemm(stream, axemb_b, axinw_b + (size_t)2 * HH, axinb + 2 * cH, nullptr, 0, vpa, cH, cS, cH, cH, 1.0f, 0);

  // 3) state init: steps[0] = conjecture
  tp_copy_f32<<<256, 256, 0, stream>>>(conj, outp, BH);
  tp_cast_f2bf<<<256, 256, 0, stream>>>(conj, state_b, BH);

  // 4) recurrent steps
  for (int t = 0; t < cSTEPS; t++) {
    // projections of current state (q pre-scaled by 1/sqrt(dh))
    gemm(stream, state_b, axinw_b,                  axinb,          nullptr, 0, q_b,  cH, cB, cH, cH, cSCALE, 0);
    gemm(stream, state_b, stinw_b,                  stinb,          nullptr, 0, qs_b, cH, cB, cH, cH, cSCALE, 0);
    gemm(stream, state_b, stinw_b + (size_t)HH,     stinb + cH,     nullptr, 0, kcache + (size_t)t * BH, cH, cB, cH, cH, 1.0f, 0);
    gemm(stream, state_b, stinw_b + (size_t)2 * HH, stinb + 2 * cH, nullptr, 0, vcache + (size_t)t * BH, cH, cB, cH, cH, 1.0f, 0);

    // comb[:, 0:1024] = state
    tp_copy_u16_strided<<<256, 256, 0, stream>>>(state_b, comb_b, cB, cH, 3072);

    // axiom attention (flash, WMMA + async LDS staging) -> ax_out into comb[:,1024:2048]
    tp_flash_axiom<<<dim3(cSPLITS, cNH, cB / 16), 32, 0, stream>>>(q_b, kpa, vpa, Opart, Mpart, Lpart);
    tp_flash_combine<<<dim3(cB, cNH), 128, 0, stream>>>(Opart, Mpart, Lpart, attn_b);
    gemm(stream, attn_b, axoutw_b, axoutb, nullptr, 0, comb_b + 1024, 3072, cB, cH, cH, 1.0f, 0);

    // proof-step attention -> st_out projection into comb[:,2048:3072]
    tp_step_attn<<<dim3(cB, cNH), 128, 0, stream>>>(qs_b, kcache, vcache, t + 1, statt_b);
    gemm(stream, statt_b, stoutw_b, stoutb, nullptr, 0, comb_b + 2048, 3072, cB, cH, cH, 1.0f, 0);

    // generator MLP: Linear(3072->2048) -> LN -> GELU -> Linear(2048->1024)
    gemm(stream, comb_b, genw1_b, genb1, x_f, 2048, nullptr, 0, cB, 2048, 3072, 1.0f, 0);
    tp_ln_gelu<<<cB, 256, 0, stream>>>(x_f, lng, lnb, x_b);
    gemm(stream, x_b, genw2_b, genb2,
         outp + (size_t)(t + 1) * BH, cH,   // steps[t+1] (fp32 output)
         state_b, cH,                       // bf16 state for next step
         cB, cH, 2048, 1.0f, 0);
  }

  // 5) validity head
  gemm(stream, state_b, valw1_b, valb1, nullptr, 0, valh_b, cH, cB, cH, cH, 1.0f, 1 /*gelu*/);
  tp_val_head<<<cB, 256, 0, stream>>>(valh_b, valw2, valb2, outp + (size_t)(cSTEPS + 1) * BH);
}
